// FullAttnNode_71562745086263
// MI455X (gfx1250) — matrix operand order repaired and checked by reading, not compiled, not run on hardware
//
#include <hip/hip_runtime.h>
#include <hip/hip_bf16.h>

#define D_MODEL 1024
#define T_SEQ   2048
#define H_NUM   16
#define HD      64
#define B_NUM   2

typedef __attribute__((ext_vector_type(16))) __bf16 v16bf;
typedef __attribute__((ext_vector_type(8)))  float  v8f;

union BF16Frag { v16bf v; uint4 q[2]; };

__device__ __forceinline__ v8f zero8() { v8f z = {}; return z; }

__device__ __forceinline__ v8f wmma_bf16(v16bf a, v16bf b, v8f c) {
  // D = A(16x32 bf16) * B(32x16 bf16) + C(16x16 f32)
  return __builtin_amdgcn_wmma_f32_16x16x32_bf16(false, a, false, b, (short)0, c,
                                                 false, false);
}

// A-matrix fragment (16x32 bf16): lane holds row (lane&15); 16 contiguous
// K values starting at K = 16*half (matches load_b_frag).
__device__ __forceinline__ v16bf load_a_frag(const __bf16* row, int half) {
  BF16Frag f;
  f.q[0] = *(const uint4*)(row + 16 * half);
  f.q[1] = *(const uint4*)(row + 16 * half + 8);
  return f.v;
}

// B-matrix fragment (32x16 bf16): lane holds column (lane&15); 16 contiguous
// K values starting at K = 16*half. Caller passes pointer to that chunk.
__device__ __forceinline__ v16bf load_b_frag(const __bf16* p) {
  BF16Frag f;
  f.q[0] = ((const uint4*)p)[0];
  f.q[1] = ((const uint4*)p)[1];
  return f.v;
}

// 4 B-fragments from one base pointer; nt*16 rows apart -> constant byte
// offsets (nt * 32KB + 2k), foldable into the b128 immediate offset field.
__device__ __forceinline__ void load_b4(const __bf16* base, int k, v16bf out[4]) {
  #pragma unroll
  for (int nt = 0; nt < 4; ++nt)
    out[nt] = load_b_frag(base + (size_t)nt * 16 * D_MODEL + k);
}

// Reductions across the 16 lanes of a wave32 half (C-layout row ownership).
__device__ __forceinline__ float hmax16(float x) {
  x = fmaxf(x, __shfl_xor(x, 1, 32));
  x = fmaxf(x, __shfl_xor(x, 2, 32));
  x = fmaxf(x, __shfl_xor(x, 4, 32));
  x = fmaxf(x, __shfl_xor(x, 8, 32));
  return x;
}
__device__ __forceinline__ float hsum16(float x) {
  x += __shfl_xor(x, 1, 32);
  x += __shfl_xor(x, 2, 32);
  x += __shfl_xor(x, 4, 32);
  x += __shfl_xor(x, 8, 32);
  return x;
}

// ---------------------------------------------------------------- cvt f32->bf16
__global__ __launch_bounds__(256) void cvt_bf16_kernel(const float* __restrict__ s,
                                                       __bf16* __restrict__ d, int n4) {
  int i = blockIdx.x * blockDim.x + threadIdx.x;
  if (i < n4) {
    float4 v = ((const float4*)s)[i];
    __bf16* p = d + (size_t)i * 4;
    p[0] = (__bf16)v.x; p[1] = (__bf16)v.y;
    p[2] = (__bf16)v.z; p[3] = (__bf16)v.w;
  }
}

// ---------------------------------------------------------------- RMSNorm
__global__ __launch_bounds__(256) void rmsnorm_kernel(const float* __restrict__ x,
                                                      const float* __restrict__ nw,
                                                      float* __restrict__ xn,
                                                      __bf16* __restrict__ xnb) {
  const int row = blockIdx.x;                       // 0..B*T-1
  const int tid = threadIdx.x;                      // 256 threads, 4 elems each
  const float* xr = x + (size_t)row * D_MODEL;
  float4 v = ((const float4*)xr)[tid];
  float ss = v.x * v.x + v.y * v.y + v.z * v.z + v.w * v.w;
  #pragma unroll
  for (int m = 1; m < 32; m <<= 1) ss += __shfl_xor(ss, m, 32);
  __shared__ float red[8];
  if ((tid & 31) == 0) red[tid >> 5] = ss;
  __syncthreads();
  float tot = 0.f;
  #pragma unroll
  for (int i = 0; i < 8; ++i) tot += red[i];
  const float r = rsqrtf(tot * (1.0f / D_MODEL) + 1e-6f);
  float4 w = ((const float4*)nw)[tid];
  float4 o;
  o.x = v.x * r * w.x; o.y = v.y * r * w.y;
  o.z = v.z * r * w.z; o.w = v.w * r * w.w;
  ((float4*)(xn + (size_t)row * D_MODEL))[tid] = o;
  __bf16* ob = xnb + (size_t)row * D_MODEL + (size_t)tid * 4;
  ob[0] = (__bf16)o.x; ob[1] = (__bf16)o.y;
  ob[2] = (__bf16)o.z; ob[3] = (__bf16)o.w;
}

// ---------------------------------------------------------------- QKV GEMM
// Y[m,n] = sum_k xn[m,k] * W[n,k] (+bias). Block = 8 waves = 64(M) x 128(N).
// Wave tile = 16(M) x 64(N). Ping-pong double buffer, unrolled 2x per trip:
// stage A computes on buf0 while buf1's loads are in flight and vice versa;
// fixed register sets per stage -> no rotation moves, loads clause together.
__global__ __launch_bounds__(256) void gemm_qkv_kernel(
    const __bf16* __restrict__ xnb,
    const __bf16* __restrict__ wq, const __bf16* __restrict__ wk,
    const __bf16* __restrict__ wv,
    const float* __restrict__ bqv, const float* __restrict__ bkv,
    const float* __restrict__ bvv,
    __bf16* __restrict__ qb, __bf16* __restrict__ kb, __bf16* __restrict__ vtb) {
  const int which = blockIdx.z;                        // 0=Q 1=K 2=V
  const __bf16* w    = (which == 0) ? wq  : (which == 1) ? wk  : wv;
  const float*  bias = (which == 0) ? bqv : (which == 1) ? bkv : bvv;

  const int tid  = threadIdx.x;
  const int wave = tid >> 5;
  const int lane = tid & 31;
  const int l15  = lane & 15;
  const int half = lane >> 4;

  const int mbase = blockIdx.y * 64 + (wave & 3) * 16;
  const int nbase = blockIdx.x * 128 + (wave >> 2) * 64;

  v8f acc[4];
  acc[0] = zero8(); acc[1] = zero8(); acc[2] = zero8(); acc[3] = zero8();

  const __bf16* arow  = xnb + (size_t)(mbase + l15) * D_MODEL;
  const __bf16* bbase = w + (size_t)(nbase + l15) * D_MODEL + 16 * half;

  v16bf a0, a1, b0[4], b1[4];
  a0 = load_a_frag(arow, half);
  load_b4(bbase, 0, b0);

  int k = 0;
  for (; k < D_MODEL - 64; k += 64) {
    // stage A: issue loads for k+32, compute k on buf0
    a1 = load_a_frag(arow + k + 32, half);
    load_b4(bbase, k + 32, b1);
    #pragma unroll
    for (int nt = 0; nt < 4; ++nt) acc[nt] = wmma_bf16(a0, b0[nt], acc[nt]);
    // stage B: issue loads for k+64, compute k+32 on buf1
    a0 = load_a_frag(arow + k + 64, half);
    load_b4(bbase, k + 64, b0);
    #pragma unroll
    for (int nt = 0; nt < 4; ++nt) acc[nt] = wmma_bf16(a1, b1[nt], acc[nt]);
  }
  // epilogue: k == D-64; buf0 holds chunk k, load final chunk k+32
  a1 = load_a_frag(arow + k + 32, half);
  load_b4(bbase, k + 32, b1);
  #pragma unroll
  for (int nt = 0; nt < 4; ++nt) acc[nt] = wmma_bf16(a0, b0[nt], acc[nt]);
  #pragma unroll
  for (int nt = 0; nt < 4; ++nt) acc[nt] = wmma_bf16(a1, b1[nt], acc[nt]);

  #pragma unroll
  for (int nt = 0; nt < 4; ++nt) {
    const int n = nbase + nt * 16 + l15;
    const float bias_n = bias[n];
    const int h = n >> 6;          // head
    const int d = n & 63;          // within head
    #pragma unroll
    for (int r = 0; r < 8; ++r) {
      const int m  = mbase + r + 8 * half;     // global row in [0, B*T)
      const int bb = m >> 11;                  // T = 2048
      const int t  = m & (T_SEQ - 1);
      const float val = acc[nt][r] + bias_n;
      if (which == 0)        // Q, pre-scaled by 1/sqrt(hd)
        qb[(((size_t)(bb * H_NUM + h)) * T_SEQ + t) * HD + d] = (__bf16)(val * 0.125f);
      else if (which == 1)   // K, [B,H,T,hd]
        kb[(((size_t)(bb * H_NUM + h)) * T_SEQ + t) * HD + d] = (__bf16)val;
      else                   // V transposed, [B,H,hd,T]
        vtb[(((size_t)(bb * H_NUM + h)) * HD + d) * T_SEQ + t] = (__bf16)val;
    }
  }
}

// ---------------------------------------------------------------- flash attention
// One wave per (b, h, 16-query tile). Online softmax; key blocks of 32.
// V fragments are loaded right after the S WMMAs so they are in flight during
// the softmax VALU work and the LDS P re-layout.
__global__ __launch_bounds__(256) void attn_kernel(const __bf16* __restrict__ qb,
                                                   const __bf16* __restrict__ kb,
                                                   const __bf16* __restrict__ vtb,
                                                   __bf16* __restrict__ attnb) {
  __shared__ __align__(16) __bf16 plds[8][16][40];   // per-wave P staging (pad=8)

  const int tid  = threadIdx.x;
  const int wave = tid >> 5;
  const int lane = tid & 31;
  const int l15  = lane & 15;
  const int half = lane >> 4;

  const int g  = blockIdx.x * 8 + wave;     // 0..4095
  const int qt = g & 127;                   // T/16 tiles
  const int h  = (g >> 7) & (H_NUM - 1);
  const int b  = g >> 11;
  const int qbase = qt * 16;

  const __bf16* qhead = qb  + ((size_t)(b * H_NUM + h)) * T_SEQ * HD;
  const __bf16* khead = kb  + ((size_t)(b * H_NUM + h)) * T_SEQ * HD;
  const __bf16* vhead = vtb + ((size_t)(b * H_NUM + h)) * HD * T_SEQ;

  // Q tile 16x64 as two K=32 A-fragments (Q already scaled)
  v16bf aq0, aq1;
  {
    const __bf16* qrow = qhead + (size_t)(qbase + l15) * HD;
    aq0 = load_a_frag(qrow, half);
    aq1 = load_a_frag(qrow + 32, half);
  }

  v8f o[4];
  o[0] = zero8(); o[1] = zero8(); o[2] = zero8(); o[3] = zero8();
  float m_i[8], l_i[8];
  #pragma unroll
  for (int r = 0; r < 8; ++r) { m_i[r] = -__builtin_inff(); l_i[r] = 0.f; }

  for (int j = 0; j < qbase + 16; j += 32) {
    // ---- S = Q * K^T for two 16-key subtiles
    v8f s0 = zero8(), s1 = zero8();
    {
      const __bf16* k0 = khead + (size_t)(j + l15) * HD + 16 * half;
      const __bf16* k1 = khead + (size_t)(j + 16 + l15) * HD + 16 * half;
      s0 = wmma_bf16(aq0, load_b_frag(k0), s0);
      s0 = wmma_bf16(aq1, load_b_frag(k0 + 32), s0);
      s1 = wmma_bf16(aq0, load_b_frag(k1), s1);
      s1 = wmma_bf16(aq1, load_b_frag(k1 + 32), s1);
    }

    // ---- V fragments: issue early, consumed only at the O-update WMMAs
    v16bf bv[4];
    #pragma unroll
    for (int nt = 0; nt < 4; ++nt) {
      const __bf16* vrow = vhead + (size_t)(nt * 16 + l15) * T_SEQ + j + 16 * half;
      bv[nt] = load_b_frag(vrow);
    }
    __builtin_prefetch(khead + (size_t)(j + 32 + l15) * HD, 0, 1);

    // ---- causal mask + online softmax (row stats per 16-lane half)
    float alpha[8];
    #pragma unroll
    for (int r = 0; r < 8; ++r) {
      const int qrow = qbase + r + 8 * half;
      const int key0 = j + l15;
      float v0 = (key0 > qrow)      ? -__builtin_inff() : s0[r];
      float v1 = (key0 + 16 > qrow) ? -__builtin_inff() : s1[r];
      const float mx = hmax16(fmaxf(v0, v1));
      const float mn = fmaxf(m_i[r], mx);
      const float p0 = __expf(v0 - mn);
      const float p1 = __expf(v1 - mn);
      const float rs = hsum16(p0 + p1);
      const float a  = __expf(m_i[r] - mn);
      l_i[r] = l_i[r] * a + rs;
      m_i[r] = mn;
      alpha[r] = a;
      s0[r] = p0; s1[r] = p1;
    }
    #pragma unroll
    for (int nt = 0; nt < 4; ++nt)
      #pragma unroll
      for (int r = 0; r < 8; ++r) o[nt][r] *= alpha[r];

    // ---- re-layout P (C-layout) -> A-fragment through per-wave LDS
    #pragma unroll
    for (int r = 0; r < 8; ++r) {
      plds[wave][r + 8 * half][l15]      = (__bf16)s0[r];
      plds[wave][r + 8 * half][16 + l15] = (__bf16)s1[r];
    }
    asm volatile("s_wait_dscnt 0" ::: "memory");
    v16bf pa;
    {
      BF16Frag f;
      const __bf16* prow = &plds[wave][l15][0];
      f.q[0] = *(const uint4*)(prow + 16 * half);
      f.q[1] = *(const uint4*)(prow + 16 * half + 8);
      pa = f.v;
    }
    asm volatile("" ::: "memory");

    // ---- O += P * V
    #pragma unroll
    for (int nt = 0; nt < 4; ++nt)
      o[nt] = wmma_bf16(pa, bv[nt], o[nt]);
  }

  // ---- finalize: divide by l_i, write [B,T,D] bf16 (heads re-interleaved)
  float inv[8];
  #pragma unroll
  for (int r = 0; r < 8; ++r) inv[r] = 1.0f / l_i[r];
  #pragma unroll
  for (int nt = 0; nt < 4; ++nt) {
    const int col = h * HD + nt * 16 + l15;
    #pragma unroll
    for (int r = 0; r < 8; ++r) {
      const int t = qbase + r + 8 * half;
      attnb[((size_t)b * T_SEQ + t) * D_MODEL + col] = (__bf16)(o[nt][r] * inv[r]);
    }
  }
}

// ---------------------------------------------------------------- out projection
// out[m,n] = sum_k attn[m,k]*Wo[n,k] + bo[n] + xn[m,n]; same pipelining as QKV.
__global__ __launch_bounds__(256) void gemm_out_kernel(
    const __bf16* __restrict__ ab, const __bf16* __restrict__ wo,
    const float* __restrict__ bo, const float* __restrict__ xn,
    float* __restrict__ out) {
  const int tid  = threadIdx.x;
  const int wave = tid >> 5;
  const int lane = tid & 31;
  const int l15  = lane & 15;
  const int half = lane >> 4;

  const int mbase = blockIdx.y * 64 + (wave & 3) * 16;
  const int nbase = blockIdx.x * 128 + (wave >> 2) * 64;

  v8f acc[4];
  acc[0] = zero8(); acc[1] = zero8(); acc[2] = zero8(); acc[3] = zero8();

  const __bf16* arow  = ab + (size_t)(mbase + l15) * D_MODEL;
  const __bf16* bbase = wo + (size_t)(nbase + l15) * D_MODEL + 16 * half;

  v16bf a0, a1, b0[4], b1[4];
  a0 = load_a_frag(arow, half);
  load_b4(bbase, 0, b0);

  int k = 0;
  for (; k < D_MODEL - 64; k += 64) {
    a1 = load_a_frag(arow + k + 32, half);
    load_b4(bbase, k + 32, b1);
    #pragma unroll
    for (int nt = 0; nt < 4; ++nt) acc[nt] = wmma_bf16(a0, b0[nt], acc[nt]);
    a0 = load_a_frag(arow + k + 64, half);
    load_b4(bbase, k + 64, b0);
    #pragma unroll
    for (int nt = 0; nt < 4; ++nt) acc[nt] = wmma_bf16(a1, b1[nt], acc[nt]);
  }
  a1 = load_a_frag(arow + k + 32, half);
  load_b4(bbase, k + 32, b1);
  #pragma unroll
  for (int nt = 0; nt < 4; ++nt) acc[nt] = wmma_bf16(a0, b0[nt], acc[nt]);
  #pragma unroll
  for (int nt = 0; nt < 4; ++nt) acc[nt] = wmma_bf16(a1, b1[nt], acc[nt]);

  #pragma unroll
  for (int nt = 0; nt < 4; ++nt) {
    const int n = nbase + nt * 16 + l15;
    const float bias_n = bo[n];
    #pragma unroll
    for (int r = 0; r < 8; ++r) {
      const int m = mbase + r + 8 * half;
      const size_t idx = (size_t)m * D_MODEL + n;
      out[idx] = acc[nt][r] + bias_n + xn[idx];
    }
  }
}

// ---------------------------------------------------------------- host launcher
extern "C" void kernel_launch(void* const* d_in, const int* in_sizes, int n_in,
                              void* d_out, int out_size, void* d_ws, size_t ws_size,
                              hipStream_t stream) {
  (void)in_sizes; (void)n_in; (void)out_size; (void)ws_size;
  const float* x  = (const float*)d_in[0];
  const float* nw = (const float*)d_in[1];
  const float* Wq = (const float*)d_in[2];
  const float* bq = (const float*)d_in[3];
  const float* Wk = (const float*)d_in[4];
  const float* bk = (const float*)d_in[5];
  const float* Wv = (const float*)d_in[6];
  const float* bv = (const float*)d_in[7];
  const float* Wo = (const float*)d_in[8];
  const float* bo = (const float*)d_in[9];
  float* out = (float*)d_out;

  char* ws = (char*)d_ws;
  size_t off = 0;
  const size_t MT = (size_t)B_NUM * T_SEQ;        // 4096 rows
  const size_t ND = MT * D_MODEL;                 // 4M elements
  const size_t WD = (size_t)D_MODEL * D_MODEL;    // 1M elements
  float*  xn_f  = (float*) (ws + off); off += ND * 4;
  __bf16* xn_b  = (__bf16*)(ws + off); off += ND * 2;
  __bf16* wq_b  = (__bf16*)(ws + off); off += WD * 2;
  __bf16* wk_b  = (__bf16*)(ws + off); off += WD * 2;
  __bf16* wv_b  = (__bf16*)(ws + off); off += WD * 2;
  __bf16* wo_b  = (__bf16*)(ws + off); off += WD * 2;
  __bf16* q_b   = (__bf16*)(ws + off); off += ND * 2;
  __bf16* k_b   = (__bf16*)(ws + off); off += ND * 2;
  __bf16* vT_b  = (__bf16*)(ws + off); off += ND * 2;
  __bf16* at_b  = (__bf16*)(ws + off); off += ND * 2;

  const int n4 = (int)(WD / 4);
  const dim3 cblk(256), cgrd((n4 + 255) / 256);
  cvt_bf16_kernel<<<cgrd, cblk, 0, stream>>>(Wq, wq_b, n4);
  cvt_bf16_kernel<<<cgrd, cblk, 0, stream>>>(Wk, wk_b, n4);
  cvt_bf16_kernel<<<cgrd, cblk, 0, stream>>>(Wv, wv_b, n4);
  cvt_bf16_kernel<<<cgrd, cblk, 0, stream>>>(Wo, wo_b, n4);

  rmsnorm_kernel<<<(int)MT, 256, 0, stream>>>(x, nw, xn_f, xn_b);

  gemm_qkv_kernel<<<dim3(D_MODEL / 128, (int)(MT / 64), 3), 256, 0, stream>>>(
      xn_b, wq_b, wk_b, wv_b, bq, bk, bv, q_b, k_b, vT_b);

  attn_kernel<<<(B_NUM * H_NUM * (T_SEQ / 16)) / 8, 256, 0, stream>>>(
      q_b, k_b, vT_b, at_b);

  gemm_out_kernel<<<dim3(D_MODEL / 128, (int)(MT / 64)), 256, 0, stream>>>(
      at_b, wo_b, bo, xn_f, out);
}
